// TransformerBlock_40656160424316
// MI455X (gfx1250) — matrix operand order repaired and checked by reading, not compiled, not run on hardware
//
#include <hip/hip_runtime.h>
#include <hip/hip_bf16.h>

// ============================================================================
// Transformer block for MI455X (gfx1250, wave32, WMMA).
// All GEMMs (QKV, attention score/value, Wo, FFN 3x) run on
// v_wmma_f32_16x16x32_bf16 with f32 accumulation. Epilogues (bias/ReLU/
// residual/output-dtype) are compile-time template parameters so the stores
// are branch-free. K-loop prefetches 512B ahead via global_prefetch_b8.
// Workspace requirement: ~400 MB (weights converted to bf16 + activations).
// ============================================================================

#define SEQL   2048
#define DMODEL 2048
#define NHEADS 16
#define HEADDIM 128
#define DFF    8192

typedef __attribute__((ext_vector_type(16))) __bf16 v16bf;
typedef __attribute__((ext_vector_type(8)))  __bf16 v8bf;
typedef __attribute__((ext_vector_type(8)))  float  v8f;
typedef __attribute__((ext_vector_type(4)))  float  v4f;

// ---- helpers ---------------------------------------------------------------

static __device__ inline __bf16 f2bf(float f) {
  unsigned u = __float_as_uint(f);
  u += 0x7FFFu + ((u >> 16) & 1u);            // round-to-nearest-even
  unsigned short h = (unsigned short)(u >> 16);
  return __builtin_bit_cast(__bf16, h);
}

static __device__ inline v8f zero8() {
  v8f z = {0.f, 0.f, 0.f, 0.f, 0.f, 0.f, 0.f, 0.f};
  return z;
}

static __device__ inline v8f wmma_bf16(v16bf a, v16bf b, v8f c) {
  return __builtin_amdgcn_wmma_f32_16x16x32_bf16(
      /*neg_a=*/false, a, /*neg_b=*/false, b,
      /*c_mod=*/(short)0, c, /*reuse_a=*/false, /*reuse_b=*/false);
}

// A-matrix fragment (16x32 bf16, M x K). Per ISA layout: lane-half h holds
// K = {h*8 .. h*8+7} U {h*8+16 .. h*8+23} of row (lane&15).
static __device__ inline v16bf load_fragA(const __bf16* __restrict__ base, int ld) {
  int lane = threadIdx.x & 31;
  int hl = lane >> 4, r = lane & 15;
  const __bf16* p = base + (size_t)r * ld + hl * 8;
  v8bf lo = *(const v8bf*)p;          // K = hl*8 .. hl*8+7
  v8bf hi = *(const v8bf*)(p + 16);   // K = hl*8+16 .. hl*8+23
  v16bf out;
#pragma unroll
  for (int i = 0; i < 8; ++i) { out[i] = lo[i]; out[8 + i] = hi[i]; }
  return out;
}

// B-matrix fragment (32x16 bf16, K x N) loaded from a row-major (N x K)
// matrix W, so B[k][n] = W[n][k]. Lane-half h holds K = {h*8 .. h*8+7} U
// {h*8+16 .. h*8+23} of column N = (lane&15).
static __device__ inline v16bf load_fragB(const __bf16* __restrict__ base, int ld) {
  int lane = threadIdx.x & 31;
  int hl = lane >> 4, r = lane & 15;
  const __bf16* p = base + (size_t)r * ld + hl * 8;
  v8bf lo = *(const v8bf*)p;
  v8bf hi = *(const v8bf*)(p + 16);
  v16bf out;
#pragma unroll
  for (int i = 0; i < 8; ++i) { out[i] = lo[i]; out[8 + i] = hi[i]; }
  return out;
}

static __device__ inline float redmax16(float v) {
#pragma unroll
  for (int s = 1; s < 16; s <<= 1) v = fmaxf(v, __shfl_xor(v, s, 32));
  return v;
}
static __device__ inline float redsum16(float v) {
#pragma unroll
  for (int s = 1; s < 16; s <<= 1) v += __shfl_xor(v, s, 32);
  return v;
}

// ---- elementwise kernels ---------------------------------------------------

__global__ void cast_f32_bf16(const float* __restrict__ in,
                              __bf16* __restrict__ out, int n) {
  int i = (blockIdx.x * blockDim.x + threadIdx.x) * 4;
  if (i + 3 < n) {
    v4f v = *(const v4f*)(in + i);
    out[i + 0] = f2bf(v.x);
    out[i + 1] = f2bf(v.y);
    out[i + 2] = f2bf(v.z);
    out[i + 3] = f2bf(v.w);
  } else {
    for (; i < n; ++i) out[i] = f2bf(in[i]);
  }
}

__global__ void rmsnorm_bf16(const float* __restrict__ x,
                             const float* __restrict__ w,
                             const float* __restrict__ b,
                             __bf16* __restrict__ out) {
  int row = blockIdx.x;
  const float* xr = x + (size_t)row * DMODEL;
  float s = 0.f;
  for (int i = threadIdx.x; i < DMODEL; i += 256) { float v = xr[i]; s += v * v; }
  __shared__ float red[256];
  red[threadIdx.x] = s;
  __syncthreads();
  for (int off = 128; off > 0; off >>= 1) {
    if ((int)threadIdx.x < off) red[threadIdx.x] += red[threadIdx.x + off];
    __syncthreads();
  }
  float inv = rsqrtf(red[0] * (1.0f / DMODEL) + 1e-5f);
  for (int i = threadIdx.x; i < DMODEL; i += 256)
    out[(size_t)row * DMODEL + i] = f2bf(xr[i] * inv * w[i] + b[i]);
}

// RoPE (equinox half-split, theta=1e4) on f32 q/k, output bf16.
__global__ void rope_cast(const float* __restrict__ qf,
                          const float* __restrict__ kf,
                          __bf16* __restrict__ qb, __bf16* __restrict__ kb) {
  int idx = blockIdx.x * blockDim.x + threadIdx.x;
  int i    = idx & 63;                 // rotation pair index, head_dim/2 = 64
  int head = (idx >> 6) & (NHEADS - 1);
  int pos  = idx >> 10;
  if (pos >= SEQL) return;
  float freq = __powf(10000.f, -(float)(2 * i) * (1.0f / HEADDIM));
  float ang = (float)pos * freq;
  float sn, cs;
  __sincosf(ang, &sn, &cs);
  size_t base = (size_t)pos * DMODEL + head * HEADDIM;
  float q1 = qf[base + i], q2 = qf[base + 64 + i];
  qb[base + i]      = f2bf(q1 * cs - q2 * sn);
  qb[base + 64 + i] = f2bf(q2 * cs + q1 * sn);
  float k1 = kf[base + i], k2 = kf[base + 64 + i];
  kb[base + i]      = f2bf(k1 * cs - k2 * sn);
  kb[base + 64 + i] = f2bf(k2 * cs + k1 * sn);
}

// V (L, H*128) f32 -> Vt (H, 128, L) bf16 so P@V's B operand is row-major.
__global__ void transpose_v(const float* __restrict__ vf, __bf16* __restrict__ vt) {
  int idx = blockIdx.x * blockDim.x + threadIdx.x;
  int pos = idx & (SEQL - 1);
  int d   = (idx >> 11) & (HEADDIM - 1);
  int h   = idx >> 18;
  if (h >= NHEADS) return;
  vt[((size_t)h * HEADDIM + d) * SEQL + pos] =
      f2bf(vf[(size_t)pos * DMODEL + h * HEADDIM + d]);
}

// ---- WMMA GEMM: out = act(A @ W^T + bias + resid) --------------------------
// A: (M,K) bf16 row-major.  W: (N,K) bf16 row-major.  M,N,K multiples of 128/32.
// 4 waves/block; each wave computes a 64x64 tile = 4x4 WMMA fragments.
// Epilogue configuration is compile-time so stores are branch-free.

template <bool HAS_BIAS, bool HAS_RESID, bool RELU, bool OUTF, bool OUTB>
__global__ __launch_bounds__(128)
void gemm_bf16(const __bf16* __restrict__ A, const __bf16* __restrict__ W,
               const float* __restrict__ bias, const float* __restrict__ resid,
               float* __restrict__ outF, __bf16* __restrict__ outB,
               int M, int N, int K) {
  int wave = threadIdx.x >> 5;
  int lane = threadIdx.x & 31;
  int hl = lane >> 4, r = lane & 15;
  int row0 = blockIdx.y * 128 + (wave >> 1) * 64;
  int col0 = blockIdx.x * 128 + (wave & 1) * 64;

  v8f C[4][4];
#pragma unroll
  for (int i = 0; i < 4; ++i)
#pragma unroll
    for (int j = 0; j < 4; ++j) C[i][j] = zero8();

  for (int k0 = 0; k0 < K; k0 += 32) {
    // Prefetch the tiles 256 K-elements (512B) ahead into the cache hierarchy
    // (lanes 0..15 / 16..31 each cover one row of the 16-row fragment).
    if (k0 + 256 < K) {
#pragma unroll
      for (int t = 0; t < 4; ++t) {
        __builtin_prefetch(A + (size_t)(row0 + t * 16 + r) * K + k0 + 256, 0, 3);
        __builtin_prefetch(W + (size_t)(col0 + t * 16 + r) * K + k0 + 256, 0, 3);
      }
    }
    v16bf a[4], b[4];
#pragma unroll
    for (int i = 0; i < 4; ++i)
      a[i] = load_fragA(A + (size_t)(row0 + i * 16) * K + k0, K);
#pragma unroll
    for (int j = 0; j < 4; ++j)
      b[j] = load_fragB(W + (size_t)(col0 + j * 16) * K + k0, K);
#pragma unroll
    for (int i = 0; i < 4; ++i)
#pragma unroll
      for (int j = 0; j < 4; ++j)
        C[i][j] = wmma_bf16(a[i], b[j], C[i][j]);
  }

  // Epilogue. C/D layout: lane-half hl, element e -> row = hl*8+e, col = r.
#pragma unroll
  for (int j = 0; j < 4; ++j) {
    int col = col0 + j * 16 + r;
    float bv = HAS_BIAS ? bias[col] : 0.f;
#pragma unroll
    for (int i = 0; i < 4; ++i) {
#pragma unroll
      for (int e = 0; e < 8; ++e) {
        int row = row0 + i * 16 + hl * 8 + e;
        size_t idx = (size_t)row * N + col;
        float v = C[i][j][e] + bv;
        if (HAS_RESID) v += resid[idx];
        if (RELU)      v = fmaxf(v, 0.f);
        if (OUTF)      outF[idx] = v;
        if (OUTB)      outB[idx] = f2bf(v);
      }
    }
  }
}

// ---- Flash attention (causal, online softmax) ------------------------------
// 1 wave per (head, 16-row query tile). Keys in 32-wide chunks.

__global__ __launch_bounds__(32)
void flash_attn(const __bf16* __restrict__ Q, const __bf16* __restrict__ Kb,
                const __bf16* __restrict__ Vt, __bf16* __restrict__ O) {
  __shared__ __bf16 pbuf[16][32];    // P tile round-trip (C-layout -> A-layout)
  int h = blockIdx.x;
  int q0 = blockIdx.y * 16;
  int lane = threadIdx.x;
  int hl = lane >> 4, r = lane & 15;
  const float scale = 0.08838834764831845f;  // 1/sqrt(128)

  v16bf qf[4];
#pragma unroll
  for (int kc = 0; kc < 4; ++kc)
    qf[kc] = load_fragA(Q + (size_t)q0 * DMODEL + h * HEADDIM + kc * 32, DMODEL);

  v8f acc[8];
  float m[8], lsum[8];
#pragma unroll
  for (int d = 0; d < 8; ++d) acc[d] = zero8();
#pragma unroll
  for (int e = 0; e < 8; ++e) { m[e] = -1e30f; lsum[e] = 0.f; }

  int jend = q0 + 15;
  for (int j0 = 0; j0 <= jend; j0 += 32) {
    // S = scale * Q K^T for keys [j0, j0+32)
    v8f s0 = zero8(), s1 = zero8();
#pragma unroll
    for (int kc = 0; kc < 4; ++kc) {
      v16bf kf0 = load_fragB(Kb + (size_t)(j0)*DMODEL + h * HEADDIM + kc * 32, DMODEL);
      v16bf kf1 = load_fragB(Kb + (size_t)(j0 + 16) * DMODEL + h * HEADDIM + kc * 32, DMODEL);
      s0 = wmma_bf16(qf[kc], kf0, s0);
      s1 = wmma_bf16(qf[kc], kf1, s1);
    }

    // scale + causal mask
#pragma unroll
    for (int e = 0; e < 8; ++e) {
      int row = q0 + hl * 8 + e;
      float v0 = s0[e] * scale; if (j0 + r > row)      v0 = -1e30f;
      float v1 = s1[e] * scale; if (j0 + 16 + r > row) v1 = -1e30f;
      s0[e] = v0; s1[e] = v1;
    }

    // online softmax statistics (per-row, 16-lane reductions)
    float alpha[8];
#pragma unroll
    for (int e = 0; e < 8; ++e) {
      float cm = redmax16(fmaxf(s0[e], s1[e]));
      float mn = fmaxf(m[e], cm);
      alpha[e] = __expf(m[e] - mn);
      float p0 = __expf(s0[e] - mn);
      float p1 = __expf(s1[e] - mn);
      float rs = redsum16(p0 + p1);
      lsum[e] = lsum[e] * alpha[e] + rs;
      m[e] = mn;
      s0[e] = p0; s1[e] = p1;
    }
#pragma unroll
    for (int d = 0; d < 8; ++d)
#pragma unroll
      for (int e = 0; e < 8; ++e) acc[d][e] *= alpha[e];

    // P (f32, C-layout) -> LDS -> bf16 A-layout fragment
    __syncthreads();
#pragma unroll
    for (int e = 0; e < 8; ++e) {
      pbuf[hl * 8 + e][r]      = f2bf(s0[e]);
      pbuf[hl * 8 + e][16 + r] = f2bf(s1[e]);
    }
    __syncthreads();
    v16bf pf;
    {
      const __bf16* p = &pbuf[r][hl * 8];
      v8bf lo = *(const v8bf*)p;
      v8bf hi = *(const v8bf*)(p + 16);
#pragma unroll
      for (int i = 0; i < 8; ++i) { pf[i] = lo[i]; pf[8 + i] = hi[i]; }
    }

    // O += P @ V  (V via pre-transposed Vt so B loads are row-major)
#pragma unroll
    for (int d = 0; d < 8; ++d) {
      v16bf vfrag = load_fragB(Vt + ((size_t)h * HEADDIM + d * 16) * SEQL + j0, SEQL);
      acc[d] = wmma_bf16(pf, vfrag, acc[d]);
    }
  }

  // normalize and write (L, H*128) bf16
#pragma unroll
  for (int d = 0; d < 8; ++d) {
#pragma unroll
    for (int e = 0; e < 8; ++e) {
      int row = q0 + hl * 8 + e;
      O[(size_t)row * DMODEL + h * HEADDIM + d * 16 + r] = f2bf(acc[d][e] / lsum[e]);
    }
  }
}

// ---- host orchestration ----------------------------------------------------

extern "C" void kernel_launch(void* const* d_in, const int* in_sizes, int n_in,
                              void* d_out, int out_size, void* d_ws, size_t ws_size,
                              hipStream_t stream) {
  const float* x    = (const float*)d_in[0];
  const float* Wq   = (const float*)d_in[1];
  const float* Wk   = (const float*)d_in[2];
  const float* Wv   = (const float*)d_in[3];
  const float* Wo   = (const float*)d_in[4];
  const float* rmsw = (const float*)d_in[5];
  const float* rmsb = (const float*)d_in[6];
  const float* W1   = (const float*)d_in[7];
  const float* b1   = (const float*)d_in[8];
  const float* W2   = (const float*)d_in[9];
  const float* b2   = (const float*)d_in[10];
  const float* W3   = (const float*)d_in[11];
  const float* b3   = (const float*)d_in[12];
  float* out = (float*)d_out;

  char* p = (char*)d_ws;
  auto carve = [&](size_t bytes) -> void* {
    void* q = (void*)p;
    p += (bytes + 255) & ~(size_t)255;
    return q;
  };

  const size_t DD = (size_t)DMODEL * DMODEL;
  const size_t LD = (size_t)SEQL * DMODEL;
  const size_t LF = (size_t)SEQL * DFF;

  __bf16* Wq_b = (__bf16*)carve(DD * 2);
  __bf16* Wk_b = (__bf16*)carve(DD * 2);
  __bf16* Wv_b = (__bf16*)carve(DD * 2);
  __bf16* Wo_b = (__bf16*)carve(DD * 2);
  __bf16* W1_b = (__bf16*)carve((size_t)DFF * DMODEL * 2);
  __bf16* W2_b = (__bf16*)carve((size_t)DFF * DFF * 2);
  __bf16* W3_b = (__bf16*)carve((size_t)DMODEL * DFF * 2);
  __bf16* xn_b  = (__bf16*)carve(LD * 2);
  float*  qf    = (float*)carve(LD * 4);
  float*  kf    = (float*)carve(LD * 4);
  float*  vf    = (float*)carve(LD * 4);
  __bf16* qb    = (__bf16*)carve(LD * 2);
  __bf16* kb    = (__bf16*)carve(LD * 2);
  __bf16* vt_b  = (__bf16*)carve(LD * 2);
  __bf16* at_b  = (__bf16*)carve(LD * 2);
  float*  x1f   = (float*)carve(LD * 4);
  __bf16* xn1_b = (__bf16*)carve(LD * 2);
  __bf16* h1_b  = (__bf16*)carve(LF * 2);
  __bf16* h2_b  = (__bf16*)carve(LF * 2);

  auto cast = [&](const float* src, __bf16* dst, size_t n) {
    int blocks = (int)((n / 4 + 255) / 256);
    cast_f32_bf16<<<blocks, 256, 0, stream>>>(src, dst, (int)n);
  };

  // 1) weights -> bf16
  cast(Wq, Wq_b, DD); cast(Wk, Wk_b, DD); cast(Wv, Wv_b, DD); cast(Wo, Wo_b, DD);
  cast(W1, W1_b, (size_t)DFF * DMODEL);
  cast(W2, W2_b, (size_t)DFF * DFF);
  cast(W3, W3_b, (size_t)DMODEL * DFF);

  // 2) xn = rmsnorm(x)
  rmsnorm_bf16<<<SEQL, 256, 0, stream>>>(x, rmsw, rmsb, xn_b);

  // 3) q,k,v projections (f32 out for RoPE / transpose): plain epilogue
  {
    dim3 grid(DMODEL / 128, SEQL / 128);
    gemm_bf16<false, false, false, true, false><<<grid, 128, 0, stream>>>(
        xn_b, Wq_b, nullptr, nullptr, qf, nullptr, SEQL, DMODEL, DMODEL);
    gemm_bf16<false, false, false, true, false><<<grid, 128, 0, stream>>>(
        xn_b, Wk_b, nullptr, nullptr, kf, nullptr, SEQL, DMODEL, DMODEL);
    gemm_bf16<false, false, false, true, false><<<grid, 128, 0, stream>>>(
        xn_b, Wv_b, nullptr, nullptr, vf, nullptr, SEQL, DMODEL, DMODEL);
  }

  // 4) RoPE + cast, V transpose
  rope_cast<<<(SEQL * NHEADS * 64) / 256, 256, 0, stream>>>(qf, kf, qb, kb);
  transpose_v<<<((size_t)NHEADS * HEADDIM * SEQL) / 256, 256, 0, stream>>>(vf, vt_b);

  // 5) causal flash attention
  flash_attn<<<dim3(NHEADS, SEQL / 16), 32, 0, stream>>>(qb, kb, vt_b, at_b);

  // 6) x1 = x + attn @ Wo^T   (residual add, f32 out)
  {
    dim3 grid(DMODEL / 128, SEQL / 128);
    gemm_bf16<false, true, false, true, false><<<grid, 128, 0, stream>>>(
        at_b, Wo_b, nullptr, x, x1f, nullptr, SEQL, DMODEL, DMODEL);
  }

  // 7) FFN
  rmsnorm_bf16<<<SEQL, 256, 0, stream>>>(x1f, rmsw, rmsb, xn1_b);
  {
    dim3 g1(DFF / 128, SEQL / 128);
    gemm_bf16<true, false, true, false, true><<<g1, 128, 0, stream>>>(
        xn1_b, W1_b, b1, nullptr, nullptr, h1_b, SEQL, DFF, DMODEL);
    gemm_bf16<true, false, true, false, true><<<g1, 128, 0, stream>>>(
        h1_b, W2_b, b2, nullptr, nullptr, h2_b, SEQL, DFF, DFF);
    dim3 g3(DMODEL / 128, SEQL / 128);
    gemm_bf16<true, true, false, true, false><<<g3, 128, 0, stream>>>(
        h2_b, W3_b, b3, x1f, out, nullptr, SEQL, DMODEL, DFF);
  }
}
